// TransformerBlock_30339648979687
// MI455X (gfx1250) — matrix operand order repaired and checked by reading, not compiled, not run on hardware
//
#include <hip/hip_runtime.h>

// ---------------------------------------------------------------------------
// Feature detection (compile-safe fallbacks)
// ---------------------------------------------------------------------------
#if defined(__has_builtin)
#if __has_builtin(__builtin_amdgcn_global_load_async_to_lds_b128)
#define HAVE_ASYNC_LDS 1
#endif
#if __has_builtin(__builtin_amdgcn_s_wait_asynccnt)
#define HAVE_WAIT_ASYNC_BUILTIN 1
#endif
#endif

// ---------------------------------------------------------------------------
// Types
// ---------------------------------------------------------------------------
typedef __bf16 bf16;
typedef __attribute__((ext_vector_type(16))) __bf16 v16bf;
typedef __attribute__((ext_vector_type(8)))  __bf16 v8bf;
typedef __attribute__((ext_vector_type(8)))  float  v8f;
typedef int v4i __attribute__((vector_size(16)));

static __device__ __forceinline__ bf16 f2bf(float f) {
  // round-to-nearest-even fp32 -> bf16
  unsigned u = __float_as_uint(f);
  u += 0x7FFFu + ((u >> 16) & 1u);
  unsigned short h = (unsigned short)(u >> 16);
  bf16 b;
  __builtin_memcpy(&b, &h, 2);
  return b;
}

static __device__ __forceinline__ v16bf comb(v8bf lo, v8bf hi) {
  v16bf r;
#pragma unroll
  for (int i = 0; i < 8; ++i) { r[i] = lo[i]; r[i + 8] = hi[i]; }
  return r;
}

static __device__ __forceinline__ v8f wmma_bf16(v16bf a, v16bf b, v8f c) {
  return __builtin_amdgcn_wmma_f32_16x16x32_bf16(false, a, false, b,
                                                 (short)0, c, false, false);
}

static __device__ __forceinline__ v8bf ld8(const bf16* p) {
  return *reinterpret_cast<const v8bf*>(p);
}

// Butterfly xor-lane exchange via ds_swizzle (group-of-32 mode: and=0x1F,
// or=0, xor=MASK). Single DS op, no address VGPR setup.
template <int MASK>
static __device__ __forceinline__ float swz(float x) {
  int i = __builtin_amdgcn_ds_swizzle(__float_as_int(x), 0x1F | (MASK << 10));
  return __int_as_float(i);
}

static __device__ __forceinline__ void wait_async0() {
#if defined(HAVE_ASYNC_LDS)
#if defined(HAVE_WAIT_ASYNC_BUILTIN)
  __builtin_amdgcn_s_wait_asynccnt(0);
#else
  asm volatile("s_wait_asynccnt 0x0" ::: "memory");
#endif
#endif
}

// Async 16-byte global -> LDS copy (CDNA5 path, tracked by ASYNCcnt).
static __device__ __forceinline__ void async_copy16(const bf16* gsrc, bf16* ldst) {
#if defined(HAVE_ASYNC_LDS)
  __builtin_amdgcn_global_load_async_to_lds_b128(
      (__attribute__((address_space(1))) v4i*)(gsrc),
      (__attribute__((address_space(3))) v4i*)(ldst), 0, 0);
#else
  *reinterpret_cast<v8bf*>(ldst) = ld8(gsrc);
#endif
}

// ---------------------------------------------------------------------------
// fp32 -> bf16 conversion
// ---------------------------------------------------------------------------
__global__ __launch_bounds__(256) void cvt_f32_bf16(const float* __restrict__ in,
                                                    bf16* __restrict__ out, int n) {
  int i = blockIdx.x * 256 + threadIdx.x;
  if (i < n) out[i] = f2bf(in[i]);
}

// ---------------------------------------------------------------------------
// Tiled bf16 WMMA GEMM: C[M,N] = A[M,K] @ B[K,N] (+bias) (+gelu)
// Block tile 128x128, BK=32, 256 threads = 8 waves (2m x 4n), wave tile 64x32.
// A tile staged with async global->LDS copies (CDNA5 path); B tile staged
// transposed through VGPRs (async copy cannot transpose).
// ---------------------------------------------------------------------------
template <bool HAS_BIAS, bool GELU, bool OUT_BF16>
__global__ __launch_bounds__(256) void gemm_wmma(
    const bf16* __restrict__ A, const bf16* __restrict__ Bm,
    const float* __restrict__ bias, void* __restrict__ Cout,
    int M, int N, int K) {
  constexpr int BM = 128, BN = 128, BK = 32;
  __shared__ bf16 As[BM][BK];   // row-major [m][k]
  __shared__ bf16 Bt[BN][BK];   // transposed [n][k]

  const int tid  = threadIdx.x;
  const int wave = tid >> 5;
  const int lane = tid & 31;
  const int l16  = lane & 15;
  const int hi   = lane >> 4;
  const int wm   = wave & 1;   // 0..1  (64 rows each)
  const int wn   = wave >> 1;  // 0..3  (32 cols each)
  const int m0 = blockIdx.y * BM;
  const int n0 = blockIdx.x * BN;

  const v8f vzero = {};
  v8f acc[4][2];
#pragma unroll
  for (int i = 0; i < 4; ++i)
#pragma unroll
    for (int j = 0; j < 2; ++j) acc[i][j] = vzero;

  // staging assignment (each thread: 2 x 8-half chunks of A, 2 of B)
  const int aRow0 = tid >> 2;          // 0..63 (+64 for second slot)
  const int aKc   = (tid & 3) * 8;
  const int bK0   = tid >> 4;          // 0..15 (+16 for second slot)
  const int bNc   = (tid & 15) * 8;

  for (int k0 = 0; k0 < K; k0 += BK) {
    __syncthreads();
#pragma unroll
    for (int s = 0; s < 2; ++s) {
      const int row = aRow0 + s * 64;
      async_copy16(A + (size_t)(m0 + row) * K + k0 + aKc, &As[row][aKc]);
      const int kk = bK0 + s * 16;
      v8bf bv = ld8(Bm + (size_t)(k0 + kk) * N + n0 + bNc);
#pragma unroll
      for (int e = 0; e < 8; ++e) Bt[bNc + e][kk] = bv[e];
    }
    if (k0 + BK < K) {
      __builtin_prefetch(Bm + (size_t)(k0 + BK + bK0) * N + n0 + bNc, 0, 1);
    }
    wait_async0();
    __syncthreads();

    // B fragments: lane n = wn*32 + j*16 + l16, k contiguous (hi half -> k 16..31)
    v16bf bfrag[2];
#pragma unroll
    for (int j = 0; j < 2; ++j) {
      const bf16* p = &Bt[wn * 32 + j * 16 + l16][hi * 16];
      bfrag[j] = comb(ld8(p), ld8(p + 8));
    }
    // A fragments: lane m = wm*64 + i*16 + l16, k contiguous (hi half -> k 16..31)
#pragma unroll
    for (int i = 0; i < 4; ++i) {
      const bf16* p = &As[wm * 64 + i * 16 + l16][hi * 16];
      v16bf afrag = comb(ld8(p), ld8(p + 8));
#pragma unroll
      for (int j = 0; j < 2; ++j)
        acc[i][j] = wmma_bf16(afrag, bfrag[j], acc[i][j]);
    }
  }

  // Epilogue. C-layout: VGPR r -> row m = r + 8*hi, col n = lane%16.
#pragma unroll
  for (int i = 0; i < 4; ++i) {
#pragma unroll
    for (int j = 0; j < 2; ++j) {
      const int row0 = m0 + wm * 64 + i * 16 + hi * 8;
      const int col  = n0 + wn * 32 + j * 16 + l16;
      const float bv = HAS_BIAS ? bias[col] : 0.0f;
#pragma unroll
      for (int r = 0; r < 8; ++r) {
        float v = acc[i][j][r] + bv;
        if (GELU) v = 0.5f * v * (1.0f + erff(v * 0.70710678118654752f));
        const size_t idx = (size_t)(row0 + r) * N + col;
        if (OUT_BF16) reinterpret_cast<bf16*>(Cout)[idx] = f2bf(v);
        else          reinterpret_cast<float*>(Cout)[idx] = v;
      }
    }
  }
}

// ---------------------------------------------------------------------------
// Flash attention (bf16 Q,K,V -> bf16 O), HD=64, scale = 1/8.
// Block = (b, h, 64 query rows); 4 waves, each owns 16 query rows.
// ---------------------------------------------------------------------------
__global__ __launch_bounds__(128) void attn_wmma(
    const bf16* __restrict__ Q, const bf16* __restrict__ K,
    const bf16* __restrict__ V, bf16* __restrict__ O, int T, int D) {
  constexpr int HD = 64;
  const int tid  = threadIdx.x;
  const int wave = tid >> 5;
  const int lane = tid & 31;
  const int l16  = lane & 15;
  const int hi   = lane >> 4;
  const int h = blockIdx.y;
  const int b = blockIdx.z;
  const size_t base = ((size_t)b * T) * D + (size_t)h * HD;
  const int q0 = blockIdx.x * 64 + wave * 16;
  const float scale = 0.125f;  // 1/sqrt(64)

  __shared__ bf16 Vt[HD][32];       // V tile transposed: [d][k]
  __shared__ bf16 Pb[4][16][32];    // per-wave P tile [m][k]

  // Q fragments (registers for entire kernel): A 16x32 over d, two d-steps.
  v16bf qf[2];
  {
    const bf16* qrow = Q + base + (size_t)(q0 + l16) * D;
#pragma unroll
    for (int ds = 0; ds < 2; ++ds) {
      const bf16* p = qrow + ds * 32 + hi * 16;
      qf[ds] = comb(ld8(p), ld8(p + 8));
    }
  }

  const v8f vzero = {};
  v8f Oacc[4];
  float mrow[8], lrow[8];
#pragma unroll
  for (int t = 0; t < 4; ++t) Oacc[t] = vzero;
#pragma unroll
  for (int r = 0; r < 8; ++r) { mrow[r] = -3.0e38f; lrow[r] = 0.0f; }

  for (int kb = 0; kb < T; kb += 32) {
    __syncthreads();
    // Stage V[kb..kb+31][0..63] transposed into LDS (all 128 threads).
    for (int s = tid; s < 256; s += 128) {
      const int k = s >> 3, dc = (s & 7) * 8;
      v8bf v = ld8(V + base + (size_t)(kb + k) * D + dc);
#pragma unroll
      for (int e = 0; e < 8; ++e) Vt[dc + e][k] = v[e];
    }
    __syncthreads();

    // Scores S = Q . K^T for 32 key columns (two 16-col n-tiles), contraction over d=64.
    // K row-major is already the B-fragment layout (n = key row, k = contiguous d).
    v8f S0 = vzero, S1 = vzero;
#pragma unroll
    for (int ds = 0; ds < 2; ++ds) {
      const bf16* kp0 = K + base + (size_t)(kb + l16) * D + ds * 32 + hi * 16;
      const bf16* kp1 = kp0 + (size_t)16 * D;
      S0 = wmma_bf16(qf[ds], comb(ld8(kp0), ld8(kp0 + 8)), S0);
      S1 = wmma_bf16(qf[ds], comb(ld8(kp1), ld8(kp1 + 8)), S1);
    }

    // Online softmax. Row r+8*hi stats live replicated across the 16 lanes of a half.
    float mb[8];
#pragma unroll
    for (int r = 0; r < 8; ++r) mb[r] = fmaxf(S0[r], S1[r]) * scale;
#pragma unroll
    for (int r = 0; r < 8; ++r) {
      mb[r] = fmaxf(mb[r], swz<1>(mb[r]));
      mb[r] = fmaxf(mb[r], swz<2>(mb[r]));
      mb[r] = fmaxf(mb[r], swz<4>(mb[r]));
      mb[r] = fmaxf(mb[r], swz<8>(mb[r]));
    }

    float alpha[8], rs[8];
#pragma unroll
    for (int r = 0; r < 8; ++r) {
      const float mn = fmaxf(mrow[r], mb[r]);
      alpha[r] = __expf(mrow[r] - mn);
      mrow[r] = mn;
    }
#pragma unroll
    for (int r = 0; r < 8; ++r) {
      const float p0 = __expf(S0[r] * scale - mrow[r]);
      const float p1 = __expf(S1[r] * scale - mrow[r]);
      rs[r] = p0 + p1;
      Pb[wave][r + hi * 8][l16]      = f2bf(p0);
      Pb[wave][r + hi * 8][16 + l16] = f2bf(p1);
    }
#pragma unroll
    for (int r = 0; r < 8; ++r) {
      rs[r] += swz<1>(rs[r]);
      rs[r] += swz<2>(rs[r]);
      rs[r] += swz<4>(rs[r]);
      rs[r] += swz<8>(rs[r]);
    }
#pragma unroll
    for (int r = 0; r < 8; ++r) lrow[r] = lrow[r] * alpha[r] + rs[r];

    // Rescale running output.
#pragma unroll
    for (int t = 0; t < 4; ++t)
#pragma unroll
      for (int r = 0; r < 8; ++r) Oacc[t][r] *= alpha[r];

    // O += P @ V_tile. A = P (LDS round-trip C-layout -> A-layout), B = Vt.
    const bf16* pp = &Pb[wave][l16][hi * 16];
    v16bf pf = comb(ld8(pp), ld8(pp + 8));
#pragma unroll
    for (int t = 0; t < 4; ++t) {
      const bf16* vp = &Vt[t * 16 + l16][hi * 16];
      Oacc[t] = wmma_bf16(pf, comb(ld8(vp), ld8(vp + 8)), Oacc[t]);
    }
  }

  // Normalize and write (bf16, feeds the o-proj GEMM).
#pragma unroll
  for (int t = 0; t < 4; ++t)
#pragma unroll
    for (int r = 0; r < 8; ++r) {
      const float v = Oacc[t][r] / lrow[r];
      O[base + (size_t)(q0 + r + hi * 8) * D + t * 16 + l16] = f2bf(v);
    }
}

// ---------------------------------------------------------------------------
// Fused residual-add + LayerNorm over D=1024. One block per row, 256 threads.
// Writes f32 (for next residual) and optionally bf16 (next GEMM A operand).
// ---------------------------------------------------------------------------
__global__ __launch_bounds__(256) void ln_kernel(
    const float* __restrict__ X, const float* __restrict__ Yadd,
    const float* __restrict__ g, const float* __restrict__ be,
    float* __restrict__ Hout, bf16* __restrict__ Hb, int writeBf) {
  constexpr int D = 1024;
  const int row = blockIdx.x;
  const size_t base = (size_t)row * D;

  float loc[4];
  float s = 0.f, s2 = 0.f;
#pragma unroll
  for (int i = 0; i < 4; ++i) {
    const int c = threadIdx.x + i * 256;
    const float v = X[base + c] + Yadd[base + c];
    loc[i] = v; s += v; s2 += v * v;
  }
  s += swz<16>(s); s2 += swz<16>(s2);
  s += swz<8>(s);  s2 += swz<8>(s2);
  s += swz<4>(s);  s2 += swz<4>(s2);
  s += swz<2>(s);  s2 += swz<2>(s2);
  s += swz<1>(s);  s2 += swz<1>(s2);

  __shared__ float sh[2][8];
  const int wave = threadIdx.x >> 5, lane = threadIdx.x & 31;
  if (lane == 0) { sh[0][wave] = s; sh[1][wave] = s2; }
  __syncthreads();
  if (wave == 0) {
    s = sh[0][lane & 7]; s2 = sh[1][lane & 7];
    s += swz<4>(s); s2 += swz<4>(s2);
    s += swz<2>(s); s2 += swz<2>(s2);
    s += swz<1>(s); s2 += swz<1>(s2);
    if (lane == 0) { sh[0][0] = s; sh[1][0] = s2; }
  }
  __syncthreads();
  s = sh[0][0]; s2 = sh[1][0];
  const float mu = s * (1.0f / D);
  const float var = s2 * (1.0f / D) - mu * mu;
  const float rr = rsqrtf(var + 1e-5f);
#pragma unroll
  for (int i = 0; i < 4; ++i) {
    const int c = threadIdx.x + i * 256;
    const float v = (loc[i] - mu) * rr * g[c] + be[c];
    Hout[base + c] = v;
    if (writeBf) Hb[base + c] = f2bf(v);
  }
}

// ---------------------------------------------------------------------------
// Host launcher
// ---------------------------------------------------------------------------
extern "C" void kernel_launch(void* const* d_in, const int* in_sizes, int n_in,
                              void* d_out, int out_size, void* d_ws, size_t ws_size,
                              hipStream_t stream) {
  (void)in_sizes; (void)n_in; (void)out_size; (void)ws_size;
  constexpr int Bc = 4, T = 2048, D = 1024, DF = 4096, H = 16;
  constexpr int M = Bc * T;  // 8192 rows

  const float* x   = (const float*)d_in[0];
  const float* wq  = (const float*)d_in[1];
  const float* wk  = (const float*)d_in[2];
  const float* wv  = (const float*)d_in[3];
  const float* wo  = (const float*)d_in[4];
  const float* w1  = (const float*)d_in[5];
  const float* b1  = (const float*)d_in[6];
  const float* w2  = (const float*)d_in[7];
  const float* b2  = (const float*)d_in[8];
  const float* g1  = (const float*)d_in[9];
  const float* be1 = (const float*)d_in[10];
  const float* g2  = (const float*)d_in[11];
  const float* be2 = (const float*)d_in[12];

  char* wsp = (char*)d_ws;
  auto alloc = [&](size_t bytes) -> void* {
    void* p = (void*)wsp;
    wsp += (bytes + 255) & ~(size_t)255;
    return p;
  };

  bf16* xb   = (bf16*)alloc((size_t)M * D * 2);
  bf16* wqb  = (bf16*)alloc((size_t)D * D * 2);
  bf16* wkb  = (bf16*)alloc((size_t)D * D * 2);
  bf16* wvb  = (bf16*)alloc((size_t)D * D * 2);
  bf16* wob  = (bf16*)alloc((size_t)D * D * 2);
  bf16* w1b  = (bf16*)alloc((size_t)D * DF * 2);
  bf16* w2b  = (bf16*)alloc((size_t)DF * D * 2);
  bf16* Qb   = (bf16*)alloc((size_t)M * D * 2);
  bf16* Kb   = (bf16*)alloc((size_t)M * D * 2);
  bf16* Vb   = (bf16*)alloc((size_t)M * D * 2);
  bf16* attnb= (bf16*)alloc((size_t)M * D * 2);
  float* y   = (float*)alloc((size_t)M * D * 4);
  float* hbuf= (float*)alloc((size_t)M * D * 4);
  bf16* hb   = (bf16*)alloc((size_t)M * D * 2);
  bf16* ub   = (bf16*)alloc((size_t)M * DF * 2);
  float* ff  = (float*)alloc((size_t)M * D * 4);

  auto cvt = [&](const float* in, bf16* out, int n) {
    cvt_f32_bf16<<<(n + 255) / 256, 256, 0, stream>>>(in, out, n);
  };
  cvt(x,  xb,  M * D);
  cvt(wq, wqb, D * D);
  cvt(wk, wkb, D * D);
  cvt(wv, wvb, D * D);
  cvt(wo, wob, D * D);
  cvt(w1, w1b, D * DF);
  cvt(w2, w2b, DF * D);

  const dim3 blk(256);
  // QKV projections (bf16 out)
  gemm_wmma<false, false, true><<<dim3(D / 128, M / 128), blk, 0, stream>>>(
      xb, wqb, nullptr, Qb, M, D, D);
  gemm_wmma<false, false, true><<<dim3(D / 128, M / 128), blk, 0, stream>>>(
      xb, wkb, nullptr, Kb, M, D, D);
  gemm_wmma<false, false, true><<<dim3(D / 128, M / 128), blk, 0, stream>>>(
      xb, wvb, nullptr, Vb, M, D, D);

  // Flash attention
  attn_wmma<<<dim3(T / 64, H, Bc), dim3(128), 0, stream>>>(Qb, Kb, Vb, attnb, T, D);

  // Output projection (f32 out)
  gemm_wmma<false, false, false><<<dim3(D / 128, M / 128), blk, 0, stream>>>(
      attnb, wob, nullptr, y, M, D, D);

  // h = LN(x + attn_out)
  ln_kernel<<<M, 256, 0, stream>>>(x, y, g1, be1, hbuf, hb, 1);

  // FFN
  gemm_wmma<true, true, true><<<dim3(DF / 128, M / 128), blk, 0, stream>>>(
      hb, w1b, b1, ub, M, DF, D);
  gemm_wmma<true, false, false><<<dim3(D / 128, M / 128), blk, 0, stream>>>(
      ub, w2b, b2, ff, M, D, DF);

  // out = LN(h + ff)
  ln_kernel<<<M, 256, 0, stream>>>(hbuf, ff, g2, be2, (float*)d_out, nullptr, 0);
}
